// RNNModel_91027536871575
// MI455X (gfx1250) — matrix operand order repaired and checked by reading, not compiled, not run on hardware
//
#include <hip/hip_runtime.h>

#define VSZ 32000
#define DSZ 1024
#define BSZ 8
#define TSZ 256
#define MROWS (BSZ * TSZ)   // 2048

typedef __attribute__((ext_vector_type(16))) __bf16 bf16x16;
typedef __attribute__((ext_vector_type(8)))  __bf16 bf16x8;
typedef __attribute__((ext_vector_type(8)))  float  f32x8;

union AFrag { bf16x16 v; bf16x8 h[2]; };

__device__ __forceinline__ unsigned short f32_to_bf16_bits(float f) {
  union { float f; unsigned u; } v; v.f = f;
  unsigned r = v.u + 0x7FFFu + ((v.u >> 16) & 1u);  // round-to-nearest-even
  return (unsigned short)(r >> 16);
}

// ---------------------------------------------------------------------------
// Generic GEMM:  C[M][N] = A[M][K] (bf16) * Bm[N][K]^T (bf16) + bias[N]
// Workgroup: 256 threads = 8 waves as 2(M) x 4(N); wave tile 64x64 (4x4 WMMA
// tiles, 16 v_wmma per K-step, 32 FLOP/B from L2), macro tile 128x256.
// ---------------------------------------------------------------------------
__global__ __launch_bounds__(256) void gemm_bf16_bias(
    const unsigned short* __restrict__ A,
    const unsigned short* __restrict__ Bm,
    const float* __restrict__ bias,
    float* __restrict__ C,
    int M, int N, int K)
{
  const int lane = threadIdx.x & 31;
  const int wave = threadIdx.x >> 5;            // 0..7
  const int half = lane >> 4;                   // 0/1
  const int r    = lane & 15;
  const int m_base = blockIdx.y * 128 + (wave >> 2) * 64;
  const int n_base = blockIdx.x * 256 + (wave & 3) * 64;

  const f32x8 zero = {};
  f32x8 acc[4][4];
  #pragma unroll
  for (int i = 0; i < 4; ++i)
    #pragma unroll
    for (int j = 0; j < 4; ++j) acc[i][j] = zero;

  for (int k0 = 0; k0 < K; k0 += 32) {
    AFrag a[4];
    #pragma unroll
    for (int i = 0; i < 4; ++i) {
      const unsigned short* p =
          A + (size_t)(m_base + i * 16 + r) * K + k0 + half * 16;
      a[i].h[0] = *reinterpret_cast<const bf16x8*>(p);
      a[i].h[1] = *reinterpret_cast<const bf16x8*>(p + 8);
    }
    #pragma unroll
    for (int j = 0; j < 4; ++j) {
      const unsigned short* p =
          Bm + (size_t)(n_base + j * 16 + r) * K + k0 + half * 16;
      const bf16x16 b = *reinterpret_cast<const bf16x16*>(p);
      __builtin_prefetch(p + 64, 0, 1);   // K-block two steps ahead -> global_prefetch
      #pragma unroll
      for (int i = 0; i < 4; ++i)
        acc[i][j] = __builtin_amdgcn_wmma_f32_16x16x32_bf16(
            false, a[i].v, false, b, (short)0, acc[i][j], false, false);
    }
  }

  #pragma unroll
  for (int j = 0; j < 4; ++j) {
    const int col  = n_base + j * 16 + r;
    const float bc = bias[col];
    #pragma unroll
    for (int i = 0; i < 4; ++i)
      #pragma unroll
      for (int e = 0; e < 8; ++e) {
        const int row = m_base + i * 16 + half * 8 + e;
        C[(size_t)row * N + col] = acc[i][j][e] + bc;
      }
  }
}

// ---------------------------------------------------------------------------
// Recurrence: h_t = tanh(Wx_t + U h_{t-1} + bU), serial in t, batch 8.
// One persistent workgroup (32 waves).  h kept in LDS as a 16x1024 bf16 tile
// (rows 8..15 permanently zero padding for the M=16 WMMA shape); row stride
// padded to 1032 shorts so the 16-lane ds_load_b128 A reads hit 16 distinct
// bank groups.  Each wave owns 32 output columns (2 WMMA tiles).
// ---------------------------------------------------------------------------
#define LDSS 1032

__global__ __launch_bounds__(1024) void rnn_recurrence(
    const unsigned short* __restrict__ Ub,  // [D][D] bf16, U[e][d]
    const float* __restrict__ Wx,           // [B*T][D]
    const float* __restrict__ bU,           // [D]
    unsigned short* __restrict__ hs)        // [B*T][D] bf16 out
{
  __shared__ unsigned short Hsh[16 * LDSS];
  for (int i = threadIdx.x; i < 16 * LDSS; i += blockDim.x) Hsh[i] = 0;
  __syncthreads();

  const int lane = threadIdx.x & 31;
  const int wave = threadIdx.x >> 5;   // 0..31
  const int half = lane >> 4;
  const int r    = lane & 15;
  const int n0   = wave * 32;

  const f32x8 zero = {};
  for (int t = 0; t < TSZ; ++t) {
    f32x8 acc0 = zero, acc1 = zero;
    for (int k0 = 0; k0 < DSZ; k0 += 32) {
      AFrag a;
      const unsigned short* p = &Hsh[r * LDSS + k0 + half * 16];
      a.h[0] = *reinterpret_cast<const bf16x8*>(p);
      a.h[1] = *reinterpret_cast<const bf16x8*>(p + 8);
      const bf16x16 b0 = *reinterpret_cast<const bf16x16*>(
          Ub + (size_t)(n0 + r) * DSZ + k0 + half * 16);
      const bf16x16 b1 = *reinterpret_cast<const bf16x16*>(
          Ub + (size_t)(n0 + 16 + r) * DSZ + k0 + half * 16);
      acc0 = __builtin_amdgcn_wmma_f32_16x16x32_bf16(
          false, a.v, false, b0, (short)0, acc0, false, false);
      acc1 = __builtin_amdgcn_wmma_f32_16x16x32_bf16(
          false, a.v, false, b1, (short)0, acc1, false, false);
    }
    __syncthreads();           // all reads of h_{t-1} done before h_t writes
    if (half == 0) {
      const int c0 = n0 + r, c1 = n0 + 16 + r;
      const float b0v = bU[c0], b1v = bU[c1];
      #pragma unroll
      for (int e = 0; e < 8; ++e) {            // e == batch index b
        const size_t rowoff = ((size_t)e * TSZ + t) * DSZ;
        const float v0 = tanhf(acc0[e] + Wx[rowoff + c0] + b0v);
        const float v1 = tanhf(acc1[e] + Wx[rowoff + c1] + b1v);
        const unsigned short q0 = f32_to_bf16_bits(v0);
        const unsigned short q1 = f32_to_bf16_bits(v1);
        Hsh[e * LDSS + c0] = q0;  Hsh[e * LDSS + c1] = q1;
        hs[rowoff + c0]    = q0;  hs[rowoff + c1]    = q1;
      }
    }
    __syncthreads();           // h_t fully visible before next step's reads
  }
}

// ---------------------------------------------------------------------------
// Helpers: f32->bf16 convert, embedding gather, online log-sum-exp, normalize
// ---------------------------------------------------------------------------
__global__ __launch_bounds__(256) void cvt_f32_bf16(
    const float* __restrict__ s, unsigned short* __restrict__ d, long n)
{
  long i = (long)blockIdx.x * blockDim.x + threadIdx.x;
  const long stride = (long)gridDim.x * blockDim.x;
  for (; i < n; i += stride) d[i] = f32_to_bf16_bits(s[i]);
}

__global__ __launch_bounds__(256) void gather_embed(
    const float* __restrict__ E, const int* __restrict__ idx,
    unsigned short* __restrict__ X)
{
  const int bt = blockIdx.x;
  const float* src = E + (size_t)idx[bt] * DSZ;
  unsigned short* dst = X + (size_t)bt * DSZ;
  for (int d = threadIdx.x; d < DSZ; d += blockDim.x)
    dst[d] = f32_to_bf16_bits(src[d]);
}

__global__ __launch_bounds__(256) void row_lse(
    const float* __restrict__ logits, float* __restrict__ lse)
{
  __shared__ float sm[256], ss[256];
  const float* p = logits + (size_t)blockIdx.x * VSZ;
  float m = -INFINITY, s = 0.0f;
  for (int c = threadIdx.x; c < VSZ; c += blockDim.x) {
    const float x = p[c];
    if (x > m) { s = s * expf(m - x) + 1.0f; m = x; }
    else       { s += expf(x - m); }
  }
  sm[threadIdx.x] = m; ss[threadIdx.x] = s;
  __syncthreads();
  for (int off = 128; off > 0; off >>= 1) {
    if (threadIdx.x < off) {
      const float m1 = sm[threadIdx.x], s1 = ss[threadIdx.x];
      const float m2 = sm[threadIdx.x + off], s2 = ss[threadIdx.x + off];
      const float mn = fmaxf(m1, m2);
      sm[threadIdx.x] = mn;
      ss[threadIdx.x] = s1 * expf(m1 - mn) + s2 * expf(m2 - mn);
    }
    __syncthreads();
  }
  if (threadIdx.x == 0) lse[blockIdx.x] = sm[0] + logf(ss[0]);
}

__global__ __launch_bounds__(256) void apply_lse(
    float* __restrict__ out, const float* __restrict__ lse)
{
  const float l = lse[blockIdx.y];
  float* p = out + (size_t)blockIdx.y * VSZ;
  int c = blockIdx.x * 1024 + threadIdx.x;
  #pragma unroll
  for (int u = 0; u < 4; ++u, c += 256)
    if (c < VSZ) p[c] -= l;
}

// ---------------------------------------------------------------------------
extern "C" void kernel_launch(void* const* d_in, const int* in_sizes, int n_in,
                              void* d_out, int out_size, void* d_ws, size_t ws_size,
                              hipStream_t stream)
{
  const int*   idx  = (const int*)  d_in[0];
  const float* E    = (const float*)d_in[1];
  const float* W    = (const float*)d_in[2];
  const float* bW   = (const float*)d_in[3];
  const float* U    = (const float*)d_in[4];
  const float* bU   = (const float*)d_in[5];
  const float* Wout = (const float*)d_in[6];
  const float* bout = (const float*)d_in[7];
  float* out = (float*)d_out;
  (void)in_sizes; (void)n_in; (void)out_size; (void)ws_size;

  char* ws = (char*)d_ws;
  size_t off = 0;
  auto alloc = [&](size_t bytes) {
    char* p = ws + off;
    off = (off + bytes + 255) & ~(size_t)255;
    return p;
  };
  unsigned short* Xb  = (unsigned short*)alloc((size_t)MROWS * DSZ * 2);
  unsigned short* Wb  = (unsigned short*)alloc((size_t)DSZ * DSZ * 2);
  unsigned short* Ub  = (unsigned short*)alloc((size_t)DSZ * DSZ * 2);
  unsigned short* Wob = (unsigned short*)alloc((size_t)VSZ * DSZ * 2);
  unsigned short* Hb  = (unsigned short*)alloc((size_t)MROWS * DSZ * 2);
  float*          Wxf = (float*)         alloc((size_t)MROWS * DSZ * 4);
  float*          lse = (float*)         alloc((size_t)MROWS * 4);

  // 1. precision conversion + embedding gather
  cvt_f32_bf16<<<1024, 256, 0, stream>>>(W,    Wb,  (long)DSZ * DSZ);
  cvt_f32_bf16<<<1024, 256, 0, stream>>>(U,    Ub,  (long)DSZ * DSZ);
  cvt_f32_bf16<<<4096, 256, 0, stream>>>(Wout, Wob, (long)VSZ * DSZ);
  gather_embed<<<MROWS, 256, 0, stream>>>(E, idx, Xb);

  // 2. Wx = X * W^T + bW   (2048 x 1024 x 1024)
  gemm_bf16_bias<<<dim3(DSZ / 256, MROWS / 128), 256, 0, stream>>>(
      Xb, Wb, bW, Wxf, MROWS, DSZ, DSZ);

  // 3. serial recurrence -> hs (bf16)
  rnn_recurrence<<<1, 1024, 0, stream>>>(Ub, Wxf, bU, Hb);

  // 4. logits = hs * Wout^T + bout   (2048 x 32000 x 1024)
  gemm_bf16_bias<<<dim3(VSZ / 256, MROWS / 128), 256, 0, stream>>>(
      Hb, Wob, bout, out, MROWS, VSZ, DSZ);

  // 5. fused log_softmax: online per-row logsumexp, then subtract
  row_lse<<<MROWS, 256, 0, stream>>>(out, lse);
  apply_lse<<<dim3(32, MROWS), 256, 0, stream>>>(out, lse);
}
